// GNNModel_47923245088957
// MI455X (gfx1250) — compile-verified, matrix operand order checked by reading, not run on hardware
//
#include <hip/hip_runtime.h>
#include <hip/hip_bf16.h>

// ---------------------------------------------------------------------------
// GNN (GCN->GAT->GCN->GAT) + fused edge-predictor MLP for MI455X (gfx1250).
// Dense projections + edge MLP run on v_wmma_f32_16x16x32_f16 (f16 in, f32 acc)
// with register double-buffering and 4 N-tiles per wave for A-fragment reuse.
// Graph scatter ops use f32 global atomics. Edge MLP fuses GEMM+bias+relu+dot.
// ---------------------------------------------------------------------------

typedef __attribute__((ext_vector_type(16))) _Float16 v16h;
typedef __attribute__((ext_vector_type(8)))  _Float16 v8h;
typedef __attribute__((ext_vector_type(8)))  float    v8f;

#define NN 50000      // nodes (multiple of 16)
#define NE 100000     // edges (multiple of 16)
#define NH 4          // attention heads
#define LEAKY 0.2f

// ---------------- WMMA fragment loads (16-bit A/B layout, wave32) -----------
// lanes 0-15 : row = lane,    K chunks {k0+0..7,  k0+16..23}
// lanes 16-31: row = lane-16, K chunks {k0+8..15, k0+24..31}
__device__ __forceinline__ v16h load_frag_ptr(const _Float16* __restrict__ p) {
  v8h lo = *(const v8h*)(p);
  v8h up = *(const v8h*)(p + 16);
  v16h f;
#pragma unroll
  for (int i = 0; i < 8; ++i) { f[i] = lo[i]; f[i + 8] = up[i]; }
  return f;
}
__device__ __forceinline__ v16h load_frag(const _Float16* __restrict__ base,
                                          int stride, int row0, int k0, int lane) {
  int r  = lane & 15;
  int hi = lane >> 4;
  return load_frag_ptr(base + (size_t)(row0 + r) * stride + k0 + (hi << 3));
}

// -------- generic WMMA GEMM: C[M,N] = A[M,K] * BT[N,K]^T --------------------
// block = (32,4): 4 waves; each wave owns TI=4 adjacent 16x16 N-tiles so the
// A fragment is reused 4x per K-step. K-loop is register double-buffered so
// the next fragment group is in flight while the current WMMAs execute.
// Requires: M % 16 == 0, N % 64 == 0, K % 32 == 0.
__global__ void k_wmma_gemm(const _Float16* __restrict__ A,
                            const _Float16* __restrict__ BT,
                            float* __restrict__ C, int M, int N, int K) {
  const int TI = 4;
  int lane = threadIdx.x;
  int m0 = blockIdx.x * 16;
  int n0 = (blockIdx.y * 4 + threadIdx.y) * (16 * TI);
  if (m0 >= M || n0 >= N) return;
  v8f acc[TI];
#pragma unroll
  for (int t = 0; t < TI; ++t) acc[t] = (v8f){};

  v16h a_cur = load_frag(A, K, m0, 0, lane);
  v16h b_cur[TI];
#pragma unroll
  for (int t = 0; t < TI; ++t) b_cur[t] = load_frag(BT, K, n0 + 16 * t, 0, lane);

  int k0 = 0;
  for (; k0 + 32 < K; k0 += 32) {
    v16h a_nxt = load_frag(A, K, m0, k0 + 32, lane);
    v16h b_nxt[TI];
#pragma unroll
    for (int t = 0; t < TI; ++t) b_nxt[t] = load_frag(BT, K, n0 + 16 * t, k0 + 32, lane);
#pragma unroll
    for (int t = 0; t < TI; ++t)
      acc[t] = __builtin_amdgcn_wmma_f32_16x16x32_f16(false, a_cur, false, b_cur[t],
                                                      (short)0, acc[t], false, false);
    a_cur = a_nxt;
#pragma unroll
    for (int t = 0; t < TI; ++t) b_cur[t] = b_nxt[t];
  }
#pragma unroll
  for (int t = 0; t < TI; ++t)
    acc[t] = __builtin_amdgcn_wmma_f32_16x16x32_f16(false, a_cur, false, b_cur[t],
                                                    (short)0, acc[t], false, false);

  // C/D layout: VGPR r -> (lanes0-15: M=r, N=lane) (lanes16-31: M=8+r, N=lane-16)
  int rb = m0 + ((lane >> 4) << 3);
#pragma unroll
  for (int t = 0; t < TI; ++t) {
    int col = n0 + 16 * t + (lane & 15);
#pragma unroll
    for (int r = 0; r < 8; ++r) C[(size_t)(rb + r) * N + col] = acc[t][r];
  }
}

// -------- fused edge predictor: sigmoid(relu([h_s|h_d] W1 + b1) . w2 + b2) --
// Each wave: 16 edges x 64 output cols; K=2048 (first 1024 from src row,
// second 1024 from dst row; 1024 % 32 == 0 so each K-block stays on one side).
// z never materialized: relu+dot(w2) in registers, shuffle-reduced, atomically
// accumulated per edge. Double-buffered like the GEMM.
__global__ void k_ep_fused(const int* __restrict__ esrc, const int* __restrict__ edst,
                           const _Float16* __restrict__ h4,   // [NN,1024] f16
                           const _Float16* __restrict__ W1T,  // [4096,2048] f16
                           const float* __restrict__ b1, const float* __restrict__ w2,
                           float* __restrict__ eacc) {
  const int C = 1024, K = 2048, TI = 4;
  int lane = threadIdx.x;
  int r = lane & 15, hi = lane >> 4;
  int e0 = blockIdx.x * 16;
  int nbase = (blockIdx.y * 4 + threadIdx.y) * (16 * TI);
  int e = e0 + r;
  const _Float16* rowS = h4 + (size_t)esrc[e] * C + (hi << 3);
  const _Float16* rowD = h4 + (size_t)edst[e] * C + (hi << 3);

  v8f acc[TI];
#pragma unroll
  for (int t = 0; t < TI; ++t) acc[t] = (v8f){};

  v16h a_cur = load_frag_ptr(rowS);
  v16h b_cur[TI];
#pragma unroll
  for (int t = 0; t < TI; ++t) b_cur[t] = load_frag(W1T, K, nbase + 16 * t, 0, lane);

  int k0 = 0;
  for (; k0 + 32 < K; k0 += 32) {
    int kn = k0 + 32;
    const _Float16* pa = (kn < C) ? (rowS + kn) : (rowD + (kn - C));
    v16h a_nxt = load_frag_ptr(pa);
    v16h b_nxt[TI];
#pragma unroll
    for (int t = 0; t < TI; ++t) b_nxt[t] = load_frag(W1T, K, nbase + 16 * t, kn, lane);
#pragma unroll
    for (int t = 0; t < TI; ++t)
      acc[t] = __builtin_amdgcn_wmma_f32_16x16x32_f16(false, a_cur, false, b_cur[t],
                                                      (short)0, acc[t], false, false);
    a_cur = a_nxt;
#pragma unroll
    for (int t = 0; t < TI; ++t) b_cur[t] = b_nxt[t];
  }
#pragma unroll
  for (int t = 0; t < TI; ++t)
    acc[t] = __builtin_amdgcn_wmma_f32_16x16x32_f16(false, a_cur, false, b_cur[t],
                                                    (short)0, acc[t], false, false);

  float rowsum[8];
#pragma unroll
  for (int rr = 0; rr < 8; ++rr) rowsum[rr] = 0.f;
#pragma unroll
  for (int t = 0; t < TI; ++t) {
    int col = nbase + t * 16 + r;
    float bb = b1[col], ww = w2[col];
#pragma unroll
    for (int rr = 0; rr < 8; ++rr) {
      float z = acc[t][rr] + bb;
      z = z > 0.f ? z : 0.f;
      rowsum[rr] += z * ww;
    }
  }
#pragma unroll
  for (int rr = 0; rr < 8; ++rr) {
    float s = rowsum[rr];
    s += __shfl_xor(s, 1, 16);
    s += __shfl_xor(s, 2, 16);
    s += __shfl_xor(s, 4, 16);
    s += __shfl_xor(s, 8, 16);
    if (r == 0) atomicAdd(&eacc[e0 + hi * 8 + rr], s);
  }
}

// ------------------------------- small kernels ------------------------------
__global__ void k_fill(float* p, float v, long n) {
  size_t i = (size_t)blockIdx.x * blockDim.x + threadIdx.x;
  size_t st = (size_t)gridDim.x * blockDim.x;
  for (; i < (size_t)n; i += st) p[i] = v;
}
__global__ void k_degree(const int* __restrict__ dst, float* deg, int E) {
  size_t i = (size_t)blockIdx.x * blockDim.x + threadIdx.x;
  size_t st = (size_t)gridDim.x * blockDim.x;
  for (; i < (size_t)E; i += st) atomicAdd(&deg[dst[i]], 1.0f);
}
__global__ void k_rsqrt_(float* d, int n) {
  size_t i = (size_t)blockIdx.x * blockDim.x + threadIdx.x;
  size_t st = (size_t)gridDim.x * blockDim.x;
  for (; i < (size_t)n; i += st) { float v = d[i]; d[i] = v > 0.f ? rsqrtf(v) : 0.f; }
}
// f32[M,K] -> f16[M,Kp] (zero pad K..Kp)
__global__ void k_cvt_pad(const float* __restrict__ s, _Float16* __restrict__ d,
                          int M, int K, int Kp) {
  size_t n = (size_t)M * Kp;
  size_t i = (size_t)blockIdx.x * blockDim.x + threadIdx.x;
  size_t st = (size_t)gridDim.x * blockDim.x;
  for (; i < n; i += st) {
    size_t m = i / Kp; int k = (int)(i % Kp);
    d[i] = (k < K) ? (_Float16)s[m * K + k] : (_Float16)0.f;
  }
}
// W[K,N] f32 -> WT[N,Kp] f16 (transpose + pad)
__global__ void k_wt_pad(const float* __restrict__ W, _Float16* __restrict__ WT,
                         int K, int N, int Kp) {
  size_t n = (size_t)N * Kp;
  size_t i = (size_t)blockIdx.x * blockDim.x + threadIdx.x;
  size_t st = (size_t)gridDim.x * blockDim.x;
  for (; i < n; i += st) {
    size_t col = i / Kp; int k = (int)(i % Kp);
    WT[i] = (k < K) ? (_Float16)W[(size_t)k * N + col] : (_Float16)0.f;
  }
}
// -------------------------------- GCN scatter -------------------------------
__global__ void k_gcn_self(const float* __restrict__ h, const float* __restrict__ dis,
                           float* __restrict__ acc, int N, int C) {
  size_t n = (size_t)N * C;
  size_t i = (size_t)blockIdx.x * blockDim.x + threadIdx.x;
  size_t st = (size_t)gridDim.x * blockDim.x;
  for (; i < n; i += st) { float w = dis[i / C]; acc[i] = w * w * h[i]; }
}
__global__ void k_gcn_edge(const int* __restrict__ src, const int* __restrict__ dst,
                           const float* __restrict__ h, const float* __restrict__ dis,
                           float* __restrict__ acc, int E, int C) {
  size_t n = (size_t)E * C;
  size_t i = (size_t)blockIdx.x * blockDim.x + threadIdx.x;
  size_t st = (size_t)gridDim.x * blockDim.x;
  for (; i < n; i += st) {
    size_t e = i / C; int c = (int)(i % C);
    int s = src[e], d = dst[e];
    atomicAdd(&acc[(size_t)d * C + c], dis[s] * dis[d] * h[(size_t)s * C + c]);
  }
}
__global__ void k_bias_relu(const float* __restrict__ a, const float* __restrict__ b,
                            float* __restrict__ o, int N, int C) {
  size_t n = (size_t)N * C;
  size_t i = (size_t)blockIdx.x * blockDim.x + threadIdx.x;
  size_t st = (size_t)gridDim.x * blockDim.x;
  for (; i < n; i += st) { float v = a[i] + b[i % C]; o[i] = v > 0.f ? v : 0.f; }
}
// -------------------------------- GAT pieces --------------------------------
__device__ __forceinline__ float leaky(float x) { return x > 0.f ? x : LEAKY * x; }
__device__ __forceinline__ void atomicMaxF(float* a, float v) {
  if (v >= 0.f) atomicMax((int*)a, __float_as_int(v));
  else          atomicMin((unsigned int*)a, __float_as_uint(v));
}
__global__ void k_gat_scores(const float* __restrict__ P, const float* __restrict__ vs,
                             const float* __restrict__ vd, float* __restrict__ a_s,
                             float* __restrict__ a_d, int N, int C) {
  size_t n = (size_t)N * NH;
  size_t i = (size_t)blockIdx.x * blockDim.x + threadIdx.x;
  size_t st = (size_t)gridDim.x * blockDim.x;
  for (; i < n; i += st) {
    size_t node = i / NH; int h = (int)(i % NH);
    const float* ph = P + (node * NH + h) * C;
    const float* ws = vs + (size_t)h * C;
    const float* wd = vd + (size_t)h * C;
    float ss = 0.f, sd = 0.f;
    for (int c = 0; c < C; ++c) { float v = ph[c]; ss += v * ws[c]; sd += v * wd[c]; }
    a_s[i] = ss; a_d[i] = sd;
  }
}
__global__ void k_gat_init_max(const float* a_s, const float* a_d, float* m, long n) {
  size_t i = (size_t)blockIdx.x * blockDim.x + threadIdx.x;
  size_t st = (size_t)gridDim.x * blockDim.x;
  for (; i < (size_t)n; i += st) m[i] = leaky(a_s[i] + a_d[i]);
}
__global__ void k_gat_edge_max(const int* __restrict__ src, const int* __restrict__ dst,
                               const float* a_s, const float* a_d, float* m, int E) {
  size_t n = (size_t)E * NH;
  size_t i = (size_t)blockIdx.x * blockDim.x + threadIdx.x;
  size_t st = (size_t)gridDim.x * blockDim.x;
  for (; i < n; i += st) {
    size_t e = i / NH; int h = (int)(i % NH);
    int s = src[e], d = dst[e];
    atomicMaxF(&m[(size_t)d * NH + h], leaky(a_s[(size_t)s * NH + h] + a_d[(size_t)d * NH + h]));
  }
}
__global__ void k_gat_init_denom(const float* a_s, const float* a_d, const float* m,
                                 float* dn, long n) {
  size_t i = (size_t)blockIdx.x * blockDim.x + threadIdx.x;
  size_t st = (size_t)gridDim.x * blockDim.x;
  for (; i < (size_t)n; i += st) dn[i] = __expf(leaky(a_s[i] + a_d[i]) - m[i]);
}
__global__ void k_gat_edge_denom(const int* __restrict__ src, const int* __restrict__ dst,
                                 const float* a_s, const float* a_d, const float* m,
                                 float* dn, int E) {
  size_t n = (size_t)E * NH;
  size_t i = (size_t)blockIdx.x * blockDim.x + threadIdx.x;
  size_t st = (size_t)gridDim.x * blockDim.x;
  for (; i < n; i += st) {
    size_t e = i / NH; int h = (int)(i % NH);
    int s = src[e], d = dst[e];
    float ev = leaky(a_s[(size_t)s * NH + h] + a_d[(size_t)d * NH + h]);
    atomicAdd(&dn[(size_t)d * NH + h], __expf(ev - m[(size_t)d * NH + h]));
  }
}
__global__ void k_gat_self_agg(const float* __restrict__ P, const float* a_s,
                               const float* a_d, const float* m, const float* dn,
                               float* __restrict__ acc, int N, int C) {
  size_t n = (size_t)N * NH * C;
  size_t i = (size_t)blockIdx.x * blockDim.x + threadIdx.x;
  size_t st = (size_t)gridDim.x * blockDim.x;
  for (; i < n; i += st) {
    size_t nh = i / C;  // node*NH + h
    float alpha = __expf(leaky(a_s[nh] + a_d[nh]) - m[nh]) / (dn[nh] + 1e-16f);
    acc[i] = alpha * P[i];
  }
}
__global__ void k_gat_edge_agg(const int* __restrict__ src, const int* __restrict__ dst,
                               const float* __restrict__ P, const float* a_s,
                               const float* a_d, const float* m, const float* dn,
                               float* __restrict__ acc, int E, int C) {
  size_t n = (size_t)E * NH;
  size_t i = (size_t)blockIdx.x * blockDim.x + threadIdx.x;
  size_t st = (size_t)gridDim.x * blockDim.x;
  for (; i < n; i += st) {
    size_t e = i / NH; int h = (int)(i % NH);
    int s = src[e], d = dst[e];
    size_t dh = (size_t)d * NH + h;
    float alpha = __expf(leaky(a_s[(size_t)s * NH + h] + a_d[dh]) - m[dh]) / (dn[dh] + 1e-16f);
    const float* ps = P + ((size_t)s * NH + h) * C;
    float* pa = acc + dh * C;
    for (int c = 0; c < C; ++c) atomicAdd(&pa[c], alpha * ps[c]);
  }
}
__global__ void k_gat_mean_bias_relu(const float* __restrict__ acc, const float* b,
                                     float* __restrict__ o, int N, int C) {
  size_t n = (size_t)N * C;
  size_t i = (size_t)blockIdx.x * blockDim.x + threadIdx.x;
  size_t st = (size_t)gridDim.x * blockDim.x;
  for (; i < n; i += st) {
    size_t node = i / C; int c = (int)(i % C);
    float s = 0.f;
    for (int h = 0; h < NH; ++h) s += acc[(node * NH + h) * C + c];
    s = s * (1.0f / NH) + b[c];
    o[i] = s > 0.f ? s : 0.f;
  }
}
__global__ void k_sigmoid(const float* eacc, const float* b2, float* out, int E) {
  size_t i = (size_t)blockIdx.x * blockDim.x + threadIdx.x;
  size_t st = (size_t)gridDim.x * blockDim.x;
  for (; i < (size_t)E; i += st) out[i] = 1.f / (1.f + __expf(-(eacc[i] + b2[0])));
}

// ------------------------------------ host ----------------------------------
static inline dim3 gs(long n) {
  long b = (n + 255) / 256;
  return dim3((unsigned)(b > 4096 ? 4096 : (b < 1 ? 1 : b)));
}

extern "C" void kernel_launch(void* const* d_in, const int* in_sizes, int n_in,
                              void* d_out, int out_size, void* d_ws, size_t ws_size,
                              hipStream_t stream) {
  (void)in_sizes; (void)n_in; (void)out_size; (void)ws_size;
  const float* x       = (const float*)d_in[0];
  const int*   ei      = (const int*)d_in[1];
  const int*   esrc    = ei;
  const int*   edst    = ei + NE;
  const float* gcn1_w  = (const float*)d_in[2];
  const float* gcn1_b  = (const float*)d_in[3];
  const float* gat1_w  = (const float*)d_in[4];
  const float* gat1_as = (const float*)d_in[5];
  const float* gat1_ad = (const float*)d_in[6];
  const float* gat1_b  = (const float*)d_in[7];
  const float* gcn2_w  = (const float*)d_in[8];
  const float* gcn2_b  = (const float*)d_in[9];
  const float* gat2_w  = (const float*)d_in[10];
  const float* gat2_as = (const float*)d_in[11];
  const float* gat2_ad = (const float*)d_in[12];
  const float* gat2_b  = (const float*)d_in[13];
  const float* ep_w1   = (const float*)d_in[14];
  const float* ep_b1   = (const float*)d_in[15];
  const float* ep_w2   = (const float*)d_in[16];
  const float* ep_b2   = (const float*)d_in[17];

  char* ws = (char*)d_ws; size_t off = 0;
  auto alloc = [&](size_t bytes) -> char* {
    char* p = ws + off; off += (bytes + 255) & ~(size_t)255; return p;
  };
  float*    dis  = (float*)alloc((size_t)NN * 4);
  float*    h1   = (float*)alloc((size_t)NN * 64 * 4);
  float*    h2   = (float*)alloc((size_t)NN * 128 * 4);
  float*    h3   = (float*)alloc((size_t)NN * 256 * 4);
  float*    h4   = (float*)alloc((size_t)NN * 1024 * 4);
  float*    a_s  = (float*)alloc((size_t)NN * NH * 4);
  float*    a_d  = (float*)alloc((size_t)NN * NH * 4);
  float*    mx   = (float*)alloc((size_t)NN * NH * 4);
  float*    dn   = (float*)alloc((size_t)NN * NH * 4);
  float*    eacc = (float*)alloc((size_t)NE * 4);
  _Float16* Af16 = (_Float16*)alloc((size_t)NN * 1024 * 2);
  _Float16* WT   = (_Float16*)alloc((size_t)4096 * 2048 * 2);
  float*    P    = (float*)alloc((size_t)NN * 4096 * 4);
  float*    ACC  = (float*)alloc((size_t)NN * 4096 * 4);

  dim3 wblk(32, 4);

  // ---- shared symmetric normalization (graph is fixed across layers) ----
  k_fill  <<<gs(NN), 256, 0, stream>>>(dis, 1.0f, NN);   // self loop
  k_degree<<<gs(NE), 256, 0, stream>>>(edst, dis, NE);
  k_rsqrt_<<<gs(NN), 256, 0, stream>>>(dis, NN);

  // ---- gcn1: [NN,16] @ [16,64] ----
  k_cvt_pad<<<gs((long)NN * 32), 256, 0, stream>>>(x, Af16, NN, 16, 32);
  k_wt_pad <<<gs(64L * 32), 256, 0, stream>>>(gcn1_w, WT, 16, 64, 32);
  k_wmma_gemm<<<dim3(NN / 16, 1), wblk, 0, stream>>>(Af16, WT, P, NN, 64, 32);
  k_gcn_self<<<gs((long)NN * 64), 256, 0, stream>>>(P, dis, ACC, NN, 64);
  k_gcn_edge<<<gs((long)NE * 64), 256, 0, stream>>>(esrc, edst, P, dis, ACC, NE, 64);
  k_bias_relu<<<gs((long)NN * 64), 256, 0, stream>>>(ACC, gcn1_b, h1, NN, 64);

  // ---- gat1: proj [NN,64] @ [64,512], H=4, C=128 ----
  k_cvt_pad<<<gs((long)NN * 64), 256, 0, stream>>>(h1, Af16, NN, 64, 64);
  k_wt_pad <<<gs(512L * 64), 256, 0, stream>>>(gat1_w, WT, 64, 512, 64);
  k_wmma_gemm<<<dim3(NN / 16, 2), wblk, 0, stream>>>(Af16, WT, P, NN, 512, 64);
  k_gat_scores<<<gs((long)NN * NH), 256, 0, stream>>>(P, gat1_as, gat1_ad, a_s, a_d, NN, 128);
  k_gat_init_max<<<gs((long)NN * NH), 256, 0, stream>>>(a_s, a_d, mx, (long)NN * NH);
  k_gat_edge_max<<<gs((long)NE * NH), 256, 0, stream>>>(esrc, edst, a_s, a_d, mx, NE);
  k_gat_init_denom<<<gs((long)NN * NH), 256, 0, stream>>>(a_s, a_d, mx, dn, (long)NN * NH);
  k_gat_edge_denom<<<gs((long)NE * NH), 256, 0, stream>>>(esrc, edst, a_s, a_d, mx, dn, NE);
  k_gat_self_agg<<<gs((long)NN * 512), 256, 0, stream>>>(P, a_s, a_d, mx, dn, ACC, NN, 128);
  k_gat_edge_agg<<<gs((long)NE * NH), 256, 0, stream>>>(esrc, edst, P, a_s, a_d, mx, dn, ACC, NE, 128);
  k_gat_mean_bias_relu<<<gs((long)NN * 128), 256, 0, stream>>>(ACC, gat1_b, h2, NN, 128);

  // ---- gcn2: [NN,128] @ [128,256] ----
  k_cvt_pad<<<gs((long)NN * 128), 256, 0, stream>>>(h2, Af16, NN, 128, 128);
  k_wt_pad <<<gs(256L * 128), 256, 0, stream>>>(gcn2_w, WT, 128, 256, 128);
  k_wmma_gemm<<<dim3(NN / 16, 1), wblk, 0, stream>>>(Af16, WT, P, NN, 256, 128);
  k_gcn_self<<<gs((long)NN * 256), 256, 0, stream>>>(P, dis, ACC, NN, 256);
  k_gcn_edge<<<gs((long)NE * 256), 256, 0, stream>>>(esrc, edst, P, dis, ACC, NE, 256);
  k_bias_relu<<<gs((long)NN * 256), 256, 0, stream>>>(ACC, gcn2_b, h3, NN, 256);

  // ---- gat2: proj [NN,256] @ [256,4096], H=4, C=1024 ----
  k_cvt_pad<<<gs((long)NN * 256), 256, 0, stream>>>(h3, Af16, NN, 256, 256);
  k_wt_pad <<<gs(4096L * 256), 256, 0, stream>>>(gat2_w, WT, 256, 4096, 256);
  k_wmma_gemm<<<dim3(NN / 16, 16), wblk, 0, stream>>>(Af16, WT, P, NN, 4096, 256);
  k_gat_scores<<<gs((long)NN * NH), 256, 0, stream>>>(P, gat2_as, gat2_ad, a_s, a_d, NN, 1024);
  k_gat_init_max<<<gs((long)NN * NH), 256, 0, stream>>>(a_s, a_d, mx, (long)NN * NH);
  k_gat_edge_max<<<gs((long)NE * NH), 256, 0, stream>>>(esrc, edst, a_s, a_d, mx, NE);
  k_gat_init_denom<<<gs((long)NN * NH), 256, 0, stream>>>(a_s, a_d, mx, dn, (long)NN * NH);
  k_gat_edge_denom<<<gs((long)NE * NH), 256, 0, stream>>>(esrc, edst, a_s, a_d, mx, dn, NE);
  k_gat_self_agg<<<gs((long)NN * 4096), 256, 0, stream>>>(P, a_s, a_d, mx, dn, ACC, NN, 1024);
  k_gat_edge_agg<<<gs((long)NE * NH), 256, 0, stream>>>(esrc, edst, P, a_s, a_d, mx, dn, ACC, NE, 1024);
  k_gat_mean_bias_relu<<<gs((long)NN * 1024), 256, 0, stream>>>(ACC, gat2_b, h4, NN, 1024);

  // ---- fused edge predictor (dominant GEMM, 1.68 TFLOP) ----
  k_cvt_pad<<<gs((long)NN * 1024), 256, 0, stream>>>(h4, Af16, NN, 1024, 1024);
  k_wt_pad <<<gs(4096L * 2048), 256, 0, stream>>>(ep_w1, WT, 2048, 4096, 2048);
  k_fill   <<<gs(NE), 256, 0, stream>>>(eacc, 0.0f, NE);
  k_ep_fused<<<dim3(NE / 16, 16), wblk, 0, stream>>>(esrc, edst, Af16, WT, ep_b1, ep_w2, eacc);
  k_sigmoid<<<gs(NE), 256, 0, stream>>>(eacc, ep_b2, (float*)d_out, NE);
}
